// CrossAttention_14705968022104
// MI455X (gfx1250) — matrix operand order repaired and checked by reading, not compiled, not run on hardware
//
#include <hip/hip_runtime.h>
#include <hip/hip_bf16.h>

typedef _Float16 half_t;
typedef __attribute__((ext_vector_type(16))) _Float16 v16h;
typedef __attribute__((ext_vector_type(8)))  float    v8f;
typedef __attribute__((ext_vector_type(4)))  unsigned int u32x4;
typedef __attribute__((ext_vector_type(8)))  int          i32x8;
typedef __attribute__((ext_vector_type(4)))  int          i32x4;

#define DIMC   1024
#define HEADS  16
#define HD     64
#define BATCH  2
#define LQ     1024
#define SKV    2048
#define SCALE_F 0.125f   // 64^-0.5

#if __has_builtin(__builtin_amdgcn_tensor_load_to_lds) && __has_builtin(__builtin_amdgcn_s_wait_tensorcnt)
#define USE_TDM 1
#else
#define USE_TDM 0
#endif

// ---------------------------------------------------------------------------
// Tiled WMMA GEMM:  C[M,N] = scale * (A[M,K] @ W[K,N]) (+ bias)
// 256 threads = 8 waves (2x4), 128x128 C tile, wave tile 64x32, K-step 32.
// A staged row-major [m][k]; W staged TRANSPOSED [n][k] so every WMMA
// fragment is two contiguous 16B LDS reads (ds_load_b128), no column walks.
// ---------------------------------------------------------------------------
#define AT_STRIDE 40    // 32 + 8 pad halves; 80B rows (16B aligned)

template<bool A_F16, bool OUT_F16>
__global__ __launch_bounds__(256)
void gemm_wmma_kernel(const void* __restrict__ Aptr, const float* __restrict__ W,
                      void* __restrict__ Cptr, const float* __restrict__ bias,
                      int M, int N, int K, float scale)
{
    __shared__ half_t lA [128 * AT_STRIDE];   // [m][k]
    __shared__ half_t lBt[128 * AT_STRIDE];   // [n][k]  (transposed W tile)

    const int tid  = threadIdx.x;
    const int lane = tid & 31;
    const int wave = tid >> 5;
    const int wrow = wave >> 2;   // 0..1
    const int wcol = wave & 3;    // 0..3
    const int lm   = lane & 15;
    const int lh   = lane >> 4;
    const int m0   = blockIdx.y * 128;
    const int n0   = blockIdx.x * 128;

    const v8f vzero = {0.f,0.f,0.f,0.f,0.f,0.f,0.f,0.f};
    v8f acc[4][2];
#pragma unroll
    for (int i = 0; i < 4; ++i)
#pragma unroll
        for (int j = 0; j < 2; ++j) acc[i][j] = vzero;

    const int arow = tid >> 1;            // 0..127
    const int aseg = (tid & 1) * 16;      // 0 / 16
    const int brow = tid >> 3;            // k row 0..31
    const int bseg = (tid & 7) * 16;      // n offset 0..112

    for (int k0 = 0; k0 < K; k0 += 32) {
        // ---- stage A tile (128 x 32) ----
        if (A_F16) {
            const half_t* ap = (const half_t*)Aptr + (size_t)(m0 + arow) * K + k0 + aseg;
            const uint4* ap4 = (const uint4*)ap;
            uint4* dst = (uint4*)&lA[arow * AT_STRIDE + aseg];
            dst[0] = ap4[0];
            dst[1] = ap4[1];
        } else {
            const float* ap = (const float*)Aptr + (size_t)(m0 + arow) * K + k0 + aseg;
#pragma unroll
            for (int i = 0; i < 16; ++i)
                lA[arow * AT_STRIDE + aseg + i] = (half_t)ap[i];
        }
        // ---- stage W tile (32 x 128) transposed -> lBt[n][k] ----
        {
            const float* bp = W + (size_t)(k0 + brow) * N + n0 + bseg;
            if (k0 + 32 < K)
                __builtin_prefetch((const void*)(bp + 32 * (size_t)N), 0, 0);
#pragma unroll
            for (int i = 0; i < 16; ++i)
                lBt[(bseg + i) * AT_STRIDE + brow] = (half_t)bp[i];
        }
        __syncthreads();

        // ---- fragments (ISA 7.12.2): all reads 2x contiguous 16B ----
        v16h afrag[4];
#pragma unroll
        for (int mt = 0; mt < 4; ++mt) {
            const half_t* base = &lA[(wrow * 64 + mt * 16 + lm) * AT_STRIDE];
#pragma unroll
            for (int e = 0; e < 16; ++e) {
                const int j = e >> 1, p = e & 1;
                const int kk = lh * 16 + e;
                afrag[mt][e] = base[kk];
            }
        }
        v16h bfrag[2];
#pragma unroll
        for (int nt = 0; nt < 2; ++nt) {
            const half_t* base = &lBt[(wcol * 32 + nt * 16 + lm) * AT_STRIDE + lh * 16];
#pragma unroll
            for (int e = 0; e < 16; ++e)
                bfrag[nt][e] = base[e];
        }

#pragma unroll
        for (int mt = 0; mt < 4; ++mt)
#pragma unroll
            for (int nt = 0; nt < 2; ++nt)
                acc[mt][nt] = __builtin_amdgcn_wmma_f32_16x16x32_f16(
                    false, afrag[mt], false, bfrag[nt],
                    (short)0, acc[mt][nt], false, false);

        __syncthreads();
    }

    // ---- store (C layout: n = lm, m = r + 8*lh) ----
#pragma unroll
    for (int mt = 0; mt < 4; ++mt)
#pragma unroll
        for (int nt = 0; nt < 2; ++nt)
#pragma unroll
            for (int r = 0; r < 8; ++r) {
                const int row = m0 + wrow * 64 + mt * 16 + r + 8 * lh;
                const int col = n0 + wcol * 32 + nt * 16 + lm;
                const float v = acc[mt][nt][r] * scale;
                if (OUT_F16)
                    ((half_t*)Cptr)[(size_t)row * N + col] = (half_t)v;
                else
                    ((float*)Cptr)[(size_t)row * N + col] = v + bias[col];
            }
}

// ---------------------------------------------------------------------------
// Flash attention: grid (L/64, B*H), 128 threads = 4 waves, wave owns 16 Q rows.
// K tiles DMA'd by the Tensor Data Mover into double-buffered LDS (TDM pads
// each 128B row by 16B -> the 72-half stride), overlapped with compute.
// V tiles staged transposed [d][s] so PV B-fragments are contiguous reads.
// ---------------------------------------------------------------------------
#define KV_STRIDE 72  // 64 + 8 pad halves; 144B rows (16B aligned)

#if USE_TDM
static __device__ __forceinline__ void tdm_load_k_tile(const half_t* gtile,
                                                       unsigned int lds_byte)
{
    const unsigned long long ga = (unsigned long long)(uintptr_t)gtile;
    // D# group 0: count=1 | lds_addr | global_addr[56:0] | type=2
    u32x4 g0 = { 1u,
                 lds_byte,
                 (unsigned int)ga,
                 (unsigned int)((ga >> 32) & 0x01FFFFFFu) | (2u << 30) };
    // D# group 1: data_size=2B, pad_en, pad_interval=128B, pad_amount=16B,
    // tensor_dim0=1024, tensor_dim1=4096, tile 64x64, dim0_stride=1024
    i32x8 g1 = { (int)((1u << 16) | (1u << 20) | (4u << 22) | (3u << 25)),
                 (int)(1024u << 16),            // tensor_dim0 low16 @ [63:48]
                 (int)(4096u << 16),            // dim0 hi16=0 | tensor_dim1 low16
                 (int)(64u << 16),              // dim1 hi16=0 | tile_dim0=64
                 (int)64u,                      // tile_dim1=64 | tile_dim2=0
                 (int)1024u,                    // tensor_dim0_stride low32
                 0, 0 };                        // stride hi / dim1_stride = 0
    i32x4 z4 = { 0, 0, 0, 0 };
    i32x8 z8 = { 0, 0, 0, 0, 0, 0, 0, 0 };
    // 6-arg toolchain variant: (g0, g1, g2, g3, extra, cpol)
    __builtin_amdgcn_tensor_load_to_lds(g0, g1, z4, z4, z8, 0);
}
#endif

__global__ __launch_bounds__(128)
void attn_wmma_kernel(const half_t* __restrict__ Q, const half_t* __restrict__ Kh,
                      const half_t* __restrict__ Vh, half_t* __restrict__ Out)
{
    __shared__ half_t lK[2][64 * KV_STRIDE];      // [s][d], double buffer
    __shared__ half_t lVt[64 * KV_STRIDE];        // [d][s], transposed
    __shared__ half_t lP[4][16][KV_STRIDE];       // per-wave P scratch

    const int tid  = threadIdx.x;
    const int lane = tid & 31;
    const int wave = tid >> 5;
    const int lm   = lane & 15;
    const int lh   = lane >> 4;
    const int bh   = blockIdx.y;
    const int b    = bh >> 4;
    const int h    = bh & 15;
    const int q0   = blockIdx.x * 64 + wave * 16;

    // Q fragments (A layout), D=64 -> two k-steps of 32
    v16h qfrag[2];
    {
        const half_t* qrow = Q + ((size_t)(b * LQ + q0 + lm)) * DIMC + h * HD;
#pragma unroll
        for (int kt = 0; kt < 2; ++kt)
#pragma unroll
            for (int e = 0; e < 16; ++e) {
                const int j = e >> 1, p = e & 1;
                const int d = kt * 32 + lh * 16 + e;
                qfrag[kt][e] = qrow[d];
            }
    }

    const v8f vzero = {0.f,0.f,0.f,0.f,0.f,0.f,0.f,0.f};
    v8f oacc[4];
#pragma unroll
    for (int nt = 0; nt < 4; ++nt) oacc[nt] = vzero;
    float mprev[8], lsum[8];
#pragma unroll
    for (int r = 0; r < 8; ++r) { mprev[r] = -3.0e38f; lsum[r] = 0.f; }

    const int srow = tid >> 1;          // 0..63
    const int dseg = (tid & 1) * 32;    // 0 / 32

#if USE_TDM
    if (wave == 0)
        tdm_load_k_tile(Kh + ((size_t)(b * SKV)) * DIMC + h * HD,
                        (unsigned int)(uintptr_t)&lK[0][0]);
#endif

    for (int sb = 0; sb < SKV / 64; ++sb) {
        const int buf = USE_TDM ? (sb & 1) : 0;

        // ---- stage V 64x64 transposed -> lVt[d][s] ----
        {
            const half_t* vp = Vh + ((size_t)(b * SKV + sb * 64 + srow)) * DIMC + h * HD + dseg;
            union { uint4 q[4]; half_t h[32]; } vb;
            const uint4* vp4 = (const uint4*)vp;
#pragma unroll
            for (int i = 0; i < 4; ++i) vb.q[i] = vp4[i];
#pragma unroll
            for (int i = 0; i < 32; ++i)
                lVt[(dseg + i) * KV_STRIDE + srow] = vb.h[i];
        }
#if USE_TDM
        if (wave == 0) __builtin_amdgcn_s_wait_tensorcnt((short)0);
#else
        {
            const half_t* kp = Kh + ((size_t)(b * SKV + sb * 64 + srow)) * DIMC + h * HD + dseg;
            const uint4* kp4 = (const uint4*)kp;
            uint4* kd = (uint4*)&lK[0][srow * KV_STRIDE + dseg];
#pragma unroll
            for (int i = 0; i < 4; ++i) kd[i] = kp4[i];
        }
#endif
        __syncthreads();
#if USE_TDM
        if (wave == 0 && sb + 1 < SKV / 64)
            tdm_load_k_tile(Kh + ((size_t)(b * SKV + (sb + 1) * 64)) * DIMC + h * HD,
                            (unsigned int)(uintptr_t)&lK[buf ^ 1][0]);
#endif

        // ---- S = Q @ K^T (16 x 64); B frag B[d][s] = K[s][d] is contiguous ----
        v8f sacc[4];
#pragma unroll
        for (int nt = 0; nt < 4; ++nt) sacc[nt] = vzero;
#pragma unroll
        for (int nt = 0; nt < 4; ++nt)
#pragma unroll
            for (int kt = 0; kt < 2; ++kt) {
                v16h bf;
                const half_t* kb = &lK[buf][(nt * 16 + lm) * KV_STRIDE + kt * 32 + lh * 16];
#pragma unroll
                for (int e = 0; e < 16; ++e) bf[e] = kb[e];
                sacc[nt] = __builtin_amdgcn_wmma_f32_16x16x32_f16(
                    false, qfrag[kt], false, bf, (short)0, sacc[nt], false, false);
            }

        // ---- online softmax (row m = r + 8*lh; reduce within 16-lane half) ----
#pragma unroll
        for (int r = 0; r < 8; ++r) {
            float mx = sacc[0][r];
#pragma unroll
            for (int nt = 1; nt < 4; ++nt) mx = fmaxf(mx, sacc[nt][r]);
#pragma unroll
            for (int off = 1; off < 16; off <<= 1)
                mx = fmaxf(mx, __shfl_xor(mx, off, 32));
            const float mnew  = fmaxf(mprev[r], mx);
            const float alpha = __expf(mprev[r] - mnew);
            float rsum = 0.f;
#pragma unroll
            for (int nt = 0; nt < 4; ++nt) {
                const float p = __expf(sacc[nt][r] - mnew);
                sacc[nt][r] = p;
                rsum += p;
            }
#pragma unroll
            for (int off = 1; off < 16; off <<= 1)
                rsum += __shfl_xor(rsum, off, 32);
            lsum[r]  = lsum[r] * alpha + rsum;
            mprev[r] = mnew;
#pragma unroll
            for (int nt = 0; nt < 4; ++nt) {
                oacc[nt][r] *= alpha;
                lP[wave][r + 8 * lh][nt * 16 + lm] = (half_t)sacc[nt][r];
            }
        }
        // same-wave LDS RAW is in-order; lP is wave-private

        // ---- O += P(16x64) @ V(64x64); B frag from lVt[d][s] contiguous ----
#pragma unroll
        for (int kt = 0; kt < 2; ++kt) {
            v16h pf;
#pragma unroll
            for (int e = 0; e < 16; ++e) {
                const int j = e >> 1, p = e & 1;
                const int s = kt * 32 + lh * 16 + e;
                pf[e] = lP[wave][lm][s];
            }
#pragma unroll
            for (int nt = 0; nt < 4; ++nt) {
                v16h vf;
                const half_t* vbp = &lVt[(nt * 16 + lm) * KV_STRIDE + kt * 32 + lh * 16];
#pragma unroll
                for (int e = 0; e < 16; ++e) vf[e] = vbp[e];
                oacc[nt] = __builtin_amdgcn_wmma_f32_16x16x32_f16(
                    false, pf, false, vf, (short)0, oacc[nt], false, false);
            }
        }
        __syncthreads();
    }

    // ---- normalize + store into head-merged [B*L, DIM] f16 ----
#pragma unroll
    for (int nt = 0; nt < 4; ++nt)
#pragma unroll
        for (int r = 0; r < 8; ++r) {
            const int row = b * LQ + q0 + r + 8 * lh;
            const int col = h * HD + nt * 16 + lm;
            Out[(size_t)row * DIMC + col] = (half_t)(oacc[nt][r] / lsum[r]);
        }
}

// ---------------------------------------------------------------------------
extern "C" void kernel_launch(void* const* d_in, const int* in_sizes, int n_in,
                              void* d_out, int out_size, void* d_ws, size_t ws_size,
                              hipStream_t stream) {
    const float* query = (const float*)d_in[0];
    const float* key   = (const float*)d_in[1];
    const float* value = (const float*)d_in[2];
    const float* Wq    = (const float*)d_in[3];
    const float* Wk    = (const float*)d_in[4];
    const float* Wv    = (const float*)d_in[5];
    const float* Wo    = (const float*)d_in[6];
    const float* bo    = (const float*)d_in[7];
    float*       out   = (float*)d_out;

    // workspace layout (f16): qf[2048x1024] kf[4096x1024] vf[4096x1024] af[2048x1024]
    char* ws = (char*)d_ws;
    half_t* qf = (half_t*)(ws);
    half_t* kf = (half_t*)(ws + (size_t)2048 * 1024 * 2);
    half_t* vf = (half_t*)(ws + (size_t)(2048 + 4096) * 1024 * 2);
    half_t* af = (half_t*)(ws + (size_t)(2048 + 4096 + 4096) * 1024 * 2);

    const dim3 blk(256);
    gemm_wmma_kernel<false, true><<<dim3(8, 16), blk, 0, stream>>>(
        query, Wq, qf, nullptr, BATCH * LQ, DIMC, DIMC, SCALE_F);
    gemm_wmma_kernel<false, true><<<dim3(8, 32), blk, 0, stream>>>(
        key, Wk, kf, nullptr, BATCH * SKV, DIMC, DIMC, 1.0f);
    gemm_wmma_kernel<false, true><<<dim3(8, 32), blk, 0, stream>>>(
        value, Wv, vf, nullptr, BATCH * SKV, DIMC, DIMC, 1.0f);
    attn_wmma_kernel<<<dim3(LQ / 64, BATCH * HEADS), dim3(128), 0, stream>>>(
        qf, kf, vf, af);
    gemm_wmma_kernel<true, false><<<dim3(8, 16), blk, 0, stream>>>(
        af, Wo, out, bo, BATCH * LQ, DIMC, DIMC, 1.0f);
}
